// mLSTMCell_39986145526477
// MI455X (gfx1250) — matrix operand order repaired and checked by reading, not compiled, not run on hardware
//
#include <hip/hip_runtime.h>
#include <hip/hip_bf16.h>
#include <math.h>

#define B_   2
#define S_   1024
#define E_   2048
#define NH_  16
#define DH_  128
#define EPS_ 1e-6f
#define SCALE_ 0.08838834764831845f   // 1/sqrt(DH)

typedef __bf16 bf16_t;
typedef __attribute__((ext_vector_type(8)))  __bf16 v8bf;
typedef __attribute__((ext_vector_type(16))) __bf16 v16bf;
typedef __attribute__((ext_vector_type(8)))  float  v8f;
typedef __attribute__((ext_vector_type(4)))  float  v4f;
typedef unsigned int u32x4 __attribute__((ext_vector_type(4)));
typedef int          i32x8 __attribute__((ext_vector_type(8)));
typedef int          i32x4 __attribute__((ext_vector_type(4)));

// ---------------------------------------------------------------------------
// Kernel 1: fp32 -> bf16 conversion of q, k, v into workspace
// ---------------------------------------------------------------------------
struct bf4 { bf16_t a, b, c, d; };

__global__ void cvt_bf16_kernel(const float* __restrict__ q,
                                const float* __restrict__ k,
                                const float* __restrict__ v,
                                bf16_t* __restrict__ qb,
                                bf16_t* __restrict__ kb,
                                bf16_t* __restrict__ vb) {
  const int which = blockIdx.y;
  const float* src = (which == 0) ? q : (which == 1) ? k : v;
  bf16_t*      dst = (which == 0) ? qb : (which == 1) ? kb : vb;
  size_t i = ((size_t)blockIdx.x * blockDim.x + threadIdx.x) * 4;
  v4f x = *(const v4f*)(src + i);
  bf4 o = { (bf16_t)x.x, (bf16_t)x.y, (bf16_t)x.z, (bf16_t)x.w };
  *(bf4*)(dst + i) = o;
}

// ---------------------------------------------------------------------------
// Kernel 2: gate pre-activations  ig/fg[b,h,s] = concat(q,k,v)[b,s,:] @ W + bias
// ---------------------------------------------------------------------------
__global__ void gates_kernel(const float* __restrict__ q,
                             const float* __restrict__ k,
                             const float* __restrict__ v,
                             const float* __restrict__ Wi,
                             const float* __restrict__ bi,
                             const float* __restrict__ Wf,
                             const float* __restrict__ bfv,
                             float* __restrict__ igA,
                             float* __restrict__ fgA) {
  const int bs = blockIdx.x;
  const int b  = bs / S_;
  const int s  = bs % S_;
  const int tid = threadIdx.x;
  const int g = tid >> 4;
  const int h = tid & 15;
  const float* qr = q + (size_t)bs * E_;
  const float* kr = k + (size_t)bs * E_;
  const float* vr = v + (size_t)bs * E_;
  float ai = 0.f, af = 0.f;
  for (int e = g; e < E_; e += 16) {
    float x = qr[e];
    ai += x * Wi[(size_t)e * NH_ + h];
    af += x * Wf[(size_t)e * NH_ + h];
  }
  for (int e = g; e < E_; e += 16) {
    float x = kr[e];
    ai += x * Wi[(size_t)(E_ + e) * NH_ + h];
    af += x * Wf[(size_t)(E_ + e) * NH_ + h];
  }
  for (int e = g; e < E_; e += 16) {
    float x = vr[e];
    ai += x * Wi[(size_t)(2 * E_ + e) * NH_ + h];
    af += x * Wf[(size_t)(2 * E_ + e) * NH_ + h];
  }
  __shared__ float sI[16][17];
  __shared__ float sF[16][17];
  sI[g][h] = ai;
  sF[g][h] = af;
  __syncthreads();
  if (tid < NH_) {
    float si = 0.f, sf = 0.f;
#pragma unroll
    for (int gg = 0; gg < 16; ++gg) { si += sI[gg][tid]; sf += sF[gg][tid]; }
    igA[((size_t)b * NH_ + tid) * S_ + s] = si + bi[tid];
    fgA[((size_t)b * NH_ + tid) * S_ + s] = sf + bfv[tid];
  }
}

// ---------------------------------------------------------------------------
// Kernel 3: per-(b,h) scans.
//   Cf[s] = inclusive cumsum of log_sigmoid(fg);  g[s] = ig[s] - Cf[s]
//   M[s]  = running max of g  (max_log_D[i] = Cf[i] + M[i])
//   n0[s] = exp(-Cf[s] - M[s]) = exp(-max_log_D)
// ---------------------------------------------------------------------------
__global__ void scan_kernel(const float* __restrict__ igA,
                            const float* __restrict__ fgA,
                            float* __restrict__ gA,
                            float* __restrict__ MA,
                            float* __restrict__ n0A) {
  const int bh = blockIdx.x;
  const int s  = threadIdx.x;
  __shared__ float buf[S_];
  const float igv = igA[(size_t)bh * S_ + s];
  const float fgv = fgA[(size_t)bh * S_ + s];
  const float lf = fminf(fgv, 0.f) - log1pf(__expf(-fabsf(fgv)));
  buf[s] = lf;
  __syncthreads();
  for (int off = 1; off < S_; off <<= 1) {
    float val = buf[s];
    if (s >= off) val += buf[s - off];
    __syncthreads();
    buf[s] = val;
    __syncthreads();
  }
  const float Cf = buf[s];
  const float gv = igv - Cf;
  __syncthreads();
  buf[s] = gv;
  __syncthreads();
  for (int off = 1; off < S_; off <<= 1) {
    float val = buf[s];
    if (s >= off) val = fmaxf(val, buf[s - off]);
    __syncthreads();
    buf[s] = val;
    __syncthreads();
  }
  const float Mv = buf[s];
  gA [(size_t)bh * S_ + s] = gv;
  MA [(size_t)bh * S_ + s] = Mv;
  n0A[(size_t)bh * S_ + s] = __expf(-Cf - Mv);
}

// ---------------------------------------------------------------------------
// Kernel 4: main flash-style mLSTM kernel (bf16 WMMA + TDM double buffering)
// ---------------------------------------------------------------------------
#define QSTR 136   // K tile LDS row stride in halfs (= 272B, TDM pad reproduces it)
#define VSTR 72    // transposed V: 128x64 bf16, +8 pad -> 144B row stride
#define PSTR 72    // per-wave P staging: 16x64 bf16, +8 pad
#define KTILE_BYTES (64 * QSTR * 2)

__device__ __forceinline__ v16bf load_fragA(const bf16_t* base, int stride, int lane) {
  const bf16_t* p = base + (lane & 15) * stride + ((lane >> 4) << 3);
  v8bf lo = *(const v8bf*)(p);
  v8bf hi = *(const v8bf*)(p + 16);
  return __builtin_shufflevector(lo, hi, 0, 1, 2, 3, 4, 5, 6, 7,
                                         8, 9, 10, 11, 12, 13, 14, 15);
}
__device__ __forceinline__ v16bf load_fragB(const bf16_t* base, int stride, int lane) {
  const bf16_t* p = base + (lane & 15) * stride + ((lane >> 4) << 3);
  v8bf lo = *(const v8bf*)(p);
  v8bf hi = *(const v8bf*)(p + 16);
  return __builtin_shufflevector(lo, hi, 0, 1, 2, 3, 4, 5, 6, 7,
                                         8, 9, 10, 11, 12, 13, 14, 15);
}

// TDM: async-DMA one 64-row x 128-elem bf16 tile (row stride E_) into LDS,
// hardware-padding +4 DWORDs after every 64 DWORDs (-> 272B LDS row stride).
__device__ __forceinline__ void tdm_load_tile(unsigned lds_off, const bf16_t* gptr) {
  const unsigned long long ga = (unsigned long long)(uintptr_t)gptr;
  u32x4 g0;
  g0[0] = 1u;                                                  // count=1
  g0[1] = lds_off;                                             // lds_addr (bytes)
  g0[2] = (unsigned)ga;                                        // global_addr lo
  g0[3] = (unsigned)((ga >> 32) & 0x01FFFFFFu) | (2u << 30);   // addr hi | type=2
  i32x8 g1;
  g1[0] = (int)((1u << 16) | (1u << 20) | (5u << 22) | (3u << 25)); // 2B elems, pad 64dw->+4dw
  g1[1] = (int)(128u << 16);                 // tensor_dim0 = 128 (lo16 in dw1[31:16])
  g1[2] = (int)(64u << 16);                  // dim0 hi16 = 0 | tensor_dim1 = 64
  g1[3] = (int)(128u << 16);                 // dim1 hi16 = 0 | tile_dim0 = 128
  g1[4] = (int)(64u);                        // tile_dim1 = 64 | tile_dim2 = 0
  g1[5] = (int)E_;                           // tensor_dim0_stride = 2048 elems
  g1[6] = 0;
  g1[7] = 0;
  const i32x4 z4 = {0, 0, 0, 0};
  const i32x8 z8 = {0, 0, 0, 0, 0, 0, 0, 0};
  // 6-arg toolchain signature: (g0, g1, g2, g3, extra, cpol)
  __builtin_amdgcn_tensor_load_to_lds(g0, g1, z4, z4, z8, 0);
}

__global__ void __launch_bounds__(128)
mlstm_main_kernel(const bf16_t* __restrict__ qb,
                  const bf16_t* __restrict__ kb,
                  const bf16_t* __restrict__ vb,
                  const float* __restrict__ gA,
                  const float* __restrict__ MA,
                  const float* __restrict__ n0A,
                  const float* __restrict__ ln_scale,
                  float* __restrict__ out) {
  __shared__ __align__(16) bf16_t Ks[2][64 * QSTR];   // TDM double buffer
  __shared__ __align__(16) bf16_t Vt[DH_ * VSTR];     // V transposed: [d][key]
  __shared__ __align__(16) bf16_t Ps[4 * 16 * PSTR];  // per-wave P staging
  __shared__ float gs[64];

  const int rowTile = blockIdx.x;  // 0..15
  const int bh      = blockIdx.y;  // 0..31
  const int b = bh >> 4, h = bh & 15;
  const int tid  = threadIdx.x;
  const int wave = tid >> 5;
  const int lane = tid & 31;
  const int col  = lane & 15;
  const int i0   = rowTile * 64;

  const size_t headOff = (size_t)b * S_ * E_ + (size_t)h * DH_;
  const unsigned ksBase = (unsigned)(uintptr_t)(&Ks[0][0]);

  // prologue: async-DMA first K tile into buffer 0
  if (wave == 0) tdm_load_tile(ksBase, kb + headOff);

  // Q A-fragments: loop-invariant, load once straight from global
  v16bf aQ[4];
  {
    const int qrow = i0 + wave * 16 + (lane & 15);
    const bf16_t* qp = qb + headOff + (size_t)qrow * E_ + ((lane >> 4) << 3);
#pragma unroll
    for (int kc = 0; kc < 4; ++kc) {
      v8bf lo = *(const v8bf*)(qp + kc * 32);
      v8bf hi = *(const v8bf*)(qp + kc * 32 + 16);
      aQ[kc] = __builtin_shufflevector(lo, hi, 0, 1, 2, 3, 4, 5, 6, 7,
                                               8, 9, 10, 11, 12, 13, 14, 15);
    }
  }

  // per-row constants (C-layout: VGPR r holds row rBase+r)
  const int rBase = i0 + wave * 16 + ((lane >> 4) << 3);
  float Mreg[8], n0reg[8], lns[8];
#pragma unroll
  for (int r = 0; r < 8; ++r) {
    Mreg[r]  = MA [(size_t)bh * S_ + rBase + r];
    n0reg[r] = n0A[(size_t)bh * S_ + rBase + r];
  }
#pragma unroll
  for (int dt = 0; dt < 8; ++dt)
    lns[dt] = ln_scale[h * DH_ + dt * 16 + col];

  const v8f zero = {0.f, 0.f, 0.f, 0.f, 0.f, 0.f, 0.f, 0.f};
  v8f hfrag[8];
#pragma unroll
  for (int dt = 0; dt < 8; ++dt) hfrag[dt] = zero;
  float sumr[8];
#pragma unroll
  for (int r = 0; r < 8; ++r) sumr[r] = 0.f;

  for (int jt = 0; jt <= rowTile; ++jt) {
    const int j0 = jt * 64;
    const int p  = jt & 1;
    __syncthreads();  // all waves done with previous Vt / K buffer reuse

    // stage V tile transposed + g values (plain loads; K arrives via TDM)
    for (int c = tid; c < 64 * 16; c += 128) {
      const int n = c >> 4, cm = c & 15;
      v8bf vv = *(const v8bf*)(vb + headOff + (size_t)(j0 + n) * E_ + cm * 8);
      const int d0 = cm * 8;
#pragma unroll
      for (int jj = 0; jj < 8; ++jj) Vt[(d0 + jj) * VSTR + n] = vv[jj];
    }
    if (tid < 64) gs[tid] = gA[(size_t)bh * S_ + j0 + tid];
    if (jt < rowTile && tid < 64)
      __builtin_prefetch(vb + headOff + (size_t)(j0 + 64 + tid) * E_, 0, 0);

    if (wave == 0) __builtin_amdgcn_s_wait_tensorcnt(0);  // K[p] DMA complete
    __syncthreads();                                      // K[p] + Vt visible

    // async-DMA next K tile into the other buffer while computing
    if (wave == 0 && jt < rowTile)
      tdm_load_tile(ksBase + (unsigned)((p ^ 1) * KTILE_BYTES),
                    kb + headOff + (size_t)(j0 + 64) * E_);

    // ---- S = Q K^T  (16x64 per wave, f32 accum) ----
    const bf16_t* kbuf = &Ks[p][0];
    v8f sfrag[4];
#pragma unroll
    for (int n = 0; n < 4; ++n) sfrag[n] = zero;
#pragma unroll
    for (int n = 0; n < 4; ++n) {
#pragma unroll
      for (int kc = 0; kc < 4; ++kc) {
        v16bf bb = load_fragB(kbuf + (n * 16) * QSTR + kc * 32, QSTR, lane);
        sfrag[n] = __builtin_amdgcn_wmma_f32_16x16x32_bf16(
            false, aQ[kc], false, bb, (short)0, sfrag[n], false, false);
      }
    }

    // ---- P = (qk/sqrt(dh)) * exp(g_j - M_i), causal mask, rowsum, stage ----
    bf16_t* Pw = &Ps[wave * 16 * PSTR];
#pragma unroll
    for (int n = 0; n < 4; ++n) {
      const int   j  = j0 + n * 16 + col;
      const float gj = gs[n * 16 + col];
#pragma unroll
      for (int r = 0; r < 8; ++r) {
        const int i = rBase + r;
        const float ex = (j <= i) ? __expf(gj - Mreg[r]) : 0.f;
        const float pv = sfrag[n][r] * SCALE_ * ex;
        sumr[r] += pv;
        Pw[(r + ((lane >> 4) << 3)) * PSTR + n * 16 + col] = (bf16_t)pv;
      }
    }
    asm volatile("s_wait_dscnt 0" ::: "memory");  // wave-local P visibility

    // ---- H += P @ V ---- (P A-fragments hoisted, reused across 8 d-tiles)
    v16bf aP[2];
#pragma unroll
    for (int kc = 0; kc < 2; ++kc) aP[kc] = load_fragA(&Pw[kc * 32], PSTR, lane);
#pragma unroll
    for (int dt = 0; dt < 8; ++dt) {
#pragma unroll
      for (int kc = 0; kc < 2; ++kc) {
        v16bf bb = load_fragB(&Vt[(dt * 16) * VSTR + kc * 32], VSTR, lane);
        hfrag[dt] = __builtin_amdgcn_wmma_f32_16x16x32_bf16(
            false, aP[kc], false, bb, (short)0, hfrag[dt], false, false);
      }
    }
  }

  // ---- normalizer: max(|rowsum|, exp(-max_log_D)) ----
  float inv[8];
#pragma unroll
  for (int r = 0; r < 8; ++r) {
    float s = sumr[r];
    s += __shfl_xor(s, 1, 32);
    s += __shfl_xor(s, 2, 32);
    s += __shfl_xor(s, 4, 32);
    s += __shfl_xor(s, 8, 32);
    inv[r] = 1.f / (fmaxf(fabsf(s), n0reg[r]) + EPS_);
  }
#pragma unroll
  for (int dt = 0; dt < 8; ++dt)
#pragma unroll
    for (int r = 0; r < 8; ++r) hfrag[dt][r] *= inv[r];

  // ---- per-head layernorm over DH=128 (half-wave xor reductions) ----
  float mu[8], rstd[8];
#pragma unroll
  for (int r = 0; r < 8; ++r) {
    float m = 0.f, q2 = 0.f;
#pragma unroll
    for (int dt = 0; dt < 8; ++dt) {
      const float x = hfrag[dt][r];
      m  += x;
      q2 += x * x;
    }
    m  += __shfl_xor(m, 1, 32);  q2 += __shfl_xor(q2, 1, 32);
    m  += __shfl_xor(m, 2, 32);  q2 += __shfl_xor(q2, 2, 32);
    m  += __shfl_xor(m, 4, 32);  q2 += __shfl_xor(q2, 4, 32);
    m  += __shfl_xor(m, 8, 32);  q2 += __shfl_xor(q2, 8, 32);
    mu[r] = m * (1.f / DH_);
    const float var = q2 * (1.f / DH_) - mu[r] * mu[r];
    rstd[r] = rsqrtf(var + EPS_);
  }

  // ---- write out[b, s, h*DH + d] ----
#pragma unroll
  for (int r = 0; r < 8; ++r) {
    const int i = rBase + r;
    float* op = out + ((size_t)b * S_ + i) * E_ + h * DH_;
#pragma unroll
    for (int dt = 0; dt < 8; ++dt)
      op[dt * 16 + col] = (hfrag[dt][r] - mu[r]) * rstd[r] * lns[dt];
  }
}

// ---------------------------------------------------------------------------
// Host launcher
// ---------------------------------------------------------------------------
extern "C" void kernel_launch(void* const* d_in, const int* in_sizes, int n_in,
                              void* d_out, int out_size, void* d_ws, size_t ws_size,
                              hipStream_t stream) {
  (void)in_sizes; (void)n_in; (void)out_size; (void)ws_size;
  const float* q   = (const float*)d_in[0];
  const float* k   = (const float*)d_in[1];
  const float* v   = (const float*)d_in[2];
  const float* Wi  = (const float*)d_in[3];
  const float* bi  = (const float*)d_in[4];
  const float* Wf  = (const float*)d_in[5];
  const float* bfv = (const float*)d_in[6];
  const float* lns = (const float*)d_in[7];
  float* out = (float*)d_out;

  char* ws = (char*)d_ws;
  const size_t NT = (size_t)B_ * S_ * E_;     // 4,194,304 elements / tensor
  const size_t NG = (size_t)B_ * NH_ * S_;    // 32,768 gate values / tensor
  bf16_t* qb  = (bf16_t*)(ws);
  bf16_t* kb  = (bf16_t*)(ws + NT * 2);
  bf16_t* vb  = (bf16_t*)(ws + NT * 4);
  float*  igA = (float*)(ws + NT * 6);
  float*  fgA = igA + NG;
  float*  gA  = fgA + NG;
  float*  MA  = gA + NG;
  float*  n0A = MA + NG;

  dim3 cg((unsigned)(NT / 4 / 256), 3);
  cvt_bf16_kernel<<<cg, 256, 0, stream>>>(q, k, v, qb, kb, vb);
  gates_kernel<<<B_ * S_, 256, 0, stream>>>(q, k, v, Wi, bi, Wf, bfv, igA, fgA);
  scan_kernel<<<B_ * NH_, S_, 0, stream>>>(igA, fgA, gA, MA, n0A);
  mlstm_main_kernel<<<dim3(S_ / 64, B_ * NH_), 128, 0, stream>>>(
      qb, kb, vb, gA, MA, n0A, lns, out);
}
